// InteractionPPBlockSUF_32384053412124
// MI455X (gfx1250) — matrix operand order repaired and checked by reading, not compiled, not run on hardware
//
#include <hip/hip_runtime.h>
#include <hip/hip_bf16.h>

typedef __bf16 bf16_t;
typedef __bf16 v16bf __attribute__((ext_vector_type(16)));
typedef float  v8f   __attribute__((ext_vector_type(8)));

__device__ __forceinline__ float silu_f(float x) {
  return x / (1.0f + __expf(-x));
}

__device__ __forceinline__ v8f vzero8() {
  v8f z;
#pragma unroll
  for (int i = 0; i < 8; ++i) z[i] = 0.0f;
  return z;
}

// Single 16x16 tile: A[16 x 32*KS] (LDS, row stride lda) x packed B ([KS][32][16]).
// A-frag (16-bit 16x32): lane half h, elem e<8 -> K = e+8h ; e>=8 -> K = 16+(e-8)+8h
// B-frag (16-bit 32x16): lane half h, elem e<8 -> K = e+8h ; e>=8 -> K = 16+(e-8)+8h ; N = lane&15
template <int KS>
__device__ __forceinline__ v8f wmma_gemm_tile(const bf16_t* A, int lda,
                                              const bf16_t* Bp, int lane, v8f acc) {
  const int h = lane >> 4;
  const int r = lane & 15;
#pragma unroll
  for (int ks = 0; ks < KS; ++ks) {
    v16bf a, b;
    const bf16_t* ar = A + r * lda + ks * 32 + 8 * h;
    const bf16_t* br = Bp + (ks * 32 + lane) * 16;
#pragma unroll
    for (int i = 0; i < 8; ++i) { a[i] = ar[i]; a[i + 8] = ar[16 + i]; }
#pragma unroll
    for (int i = 0; i < 16; ++i) b[i] = br[i];
    acc = __builtin_amdgcn_wmma_f32_16x16x32_bf16(false, a, false, b, (short)0, acc,
                                                  false, false);
  }
  return acc;
}

// MT M-tiles (rows m*16..m*16+15 of an MT*16-row LDS tile) sharing each B
// fragment: MT-way B-operand reuse -> divides L2 weight traffic per WMMA by MT.
template <int KS, int MT>
__device__ __forceinline__ void wmma_multi(const bf16_t* A, int lda, const bf16_t* Bp,
                                           int lane, v8f* c) {
  const int h = lane >> 4;
  const int r = lane & 15;
#pragma unroll
  for (int ks = 0; ks < KS; ++ks) {
    v16bf b;
    const bf16_t* br = Bp + (ks * 32 + lane) * 16;
#pragma unroll
    for (int i = 0; i < 16; ++i) b[i] = br[i];
#pragma unroll
    for (int m = 0; m < MT; ++m) {
      v16bf a;
      const bf16_t* ar = A + (m * 16 + r) * lda + ks * 32 + 8 * h;
#pragma unroll
      for (int i = 0; i < 8; ++i) { a[i] = ar[i]; a[i + 8] = ar[16 + i]; }
      c[m] = __builtin_amdgcn_wmma_f32_16x16x32_bf16(false, a, false, b, (short)0,
                                                     c[m], false, false);
    }
  }
}

// ---------------- weight packing: W[K][N] f32 -> bf16 fragment order ----------
__global__ void pack_w(const float* __restrict__ src, bf16_t* __restrict__ dst,
                       int K, int N, int batch) {
  const size_t per = (size_t)K * N;
  const size_t total = per * batch;
  const int KS = K >> 5;
  for (size_t p0 = (size_t)blockIdx.x * blockDim.x + threadIdx.x; p0 < total;
       p0 += (size_t)gridDim.x * blockDim.x) {
    size_t mat = p0 / per;
    int p = (int)(p0 - mat * per);
    int e = p & 15;
    int lane = (p >> 4) & 31;
    int rest = p >> 9;
    int ks = rest % KS;
    int nt = rest / KS;
    int hh = lane >> 4;
    int c = nt * 16 + (lane & 15);
    int k = ks * 32 + ((e < 8) ? (e + 8 * hh) : (16 + 8 * hh + (e - 8)));
    dst[p0] = (bf16_t)src[mat * per + (size_t)k * N + c];
  }
}

__global__ void cvt_bf16(const float* __restrict__ src, bf16_t* __restrict__ dst,
                         size_t n) {
  size_t i = (size_t)blockIdx.x * blockDim.x + threadIdx.x;
  if (i < n) dst[i] = (bf16_t)src[i];
}

// rbf_proj[E,128] = (rbf @ W_rbf1) @ W_rbf2   (rank-8 bottleneck, VALU is fine)
__global__ void rbfproj_kernel(const float* __restrict__ rbf,
                               const float* __restrict__ W1,  // [R,B]
                               const float* __restrict__ W2,  // [B,128]
                               float* __restrict__ rp, int E, int R, int B) {
  int e = blockIdx.x * blockDim.x + threadIdx.x;
  if (e >= E) return;
  float t[8];
#pragma unroll
  for (int j = 0; j < 8; ++j) t[j] = 0.0f;
  for (int s = 0; s < R; ++s) {
    float v = rbf[(size_t)e * R + s];
    for (int j = 0; j < B; ++j) t[j] += v * W1[s * B + j];
  }
  for (int hc = 0; hc < 128; ++hc) {
    float a = 0.0f;
    for (int j = 0; j < B; ++j) a += t[j] * W2[j * 128 + hc];
    rp[(size_t)e * 128 + hc] = a;
  }
}

// x_kj chain: silu(silu(x@W_kj + b_kj) * rbf_proj @ W_down) -> xkj[E,64]
__global__ void __launch_bounds__(256)
xkj_kernel(const bf16_t* __restrict__ xbf, const float* __restrict__ rbf_proj,
           const bf16_t* __restrict__ pk_kj,    // [8][4][32][16]
           const bf16_t* __restrict__ pk_down,  // [4][4][32][16]
           const float* __restrict__ b_kj, float* __restrict__ xkj, int E) {
  __shared__ bf16_t xs[16][136];
  const int tid = threadIdx.x, wv = tid >> 5, lane = tid & 31;
  const int hh = lane >> 4, col = wv * 16 + (lane & 15);
  const int row0 = blockIdx.x * 16;

  for (int i = tid; i < 16 * 128; i += 256) {
    int r = i >> 7, c = i & 127;
    int gr = row0 + r;
    xs[r][c] = (gr < E) ? xbf[(size_t)gr * 128 + c] : (bf16_t)0.0f;
  }
  __syncthreads();

  v8f acc = wmma_gemm_tile<4>(&xs[0][0], 136, pk_kj + (size_t)wv * 2048, lane, vzero8());
  float v[8];
  const float bk = b_kj[col];
#pragma unroll
  for (int r = 0; r < 8; ++r) {
    int gr = row0 + r + 8 * hh;
    float rp = (gr < E) ? rbf_proj[(size_t)gr * 128 + col] : 0.0f;
    v[r] = silu_f(acc[r] + bk) * rp;
  }
  __syncthreads();
#pragma unroll
  for (int r = 0; r < 8; ++r) xs[r + 8 * hh][col] = (bf16_t)v[r];
  __syncthreads();
  if (wv < 4) {  // N=64 -> 4 tiles; wave-uniform branch keeps EXEC full
    v8f a2 = wmma_gemm_tile<4>(&xs[0][0], 136, pk_down + (size_t)wv * 2048, lane, vzero8());
#pragma unroll
    for (int r = 0; r < 8; ++r) {
      int gr = row0 + r + 8 * hh;
      if (gr < E) xkj[(size_t)gr * 64 + col] = silu_f(a2[r]);
    }
  }
}

// triplets: sbf_e = (sbf@W_sbf1)@W_sbf2 ; x_kj_t = xkj[idx_kj]*sbf_e ;
// fused single scatter into agg[branch][idx_ji]  (branch = bt[idx_kj]+1)
__global__ void triplet_kernel(const float* __restrict__ sbf,
                               const float* __restrict__ Ws1,  // [SR,B2]
                               const float* __restrict__ Ws2,  // [B2,64]
                               const float* __restrict__ xkj,
                               const int* __restrict__ idx_kj,
                               const int* __restrict__ idx_ji,
                               const int* __restrict__ bt,
                               float* __restrict__ agg, int T, int E, int SR, int B2) {
  int t = blockIdx.x * blockDim.x + threadIdx.x;
  if (t >= T) return;
  float t8[8];
#pragma unroll
  for (int j = 0; j < 8; ++j) t8[j] = 0.0f;
  const float* srow = sbf + (size_t)t * SR;
  for (int s = 0; s < SR; ++s) {
    float v = srow[s];
    for (int j = 0; j < B2; ++j) t8[j] += v * Ws1[s * B2 + j];
  }
  int kj = idx_kj[t], ji = idx_ji[t];
  int br = bt[kj] + 1;
  const float* xr = xkj + (size_t)kj * 64;
  float* ar = agg + ((size_t)br * E + ji) * 64;
  for (int i = 0; i < 64; ++i) {
    float sv = 0.0f;
    for (int j = 0; j < B2; ++j) sv += t8[j] * Ws2[j * 64 + i];
    atomicAdd(&ar[i], xr[i] * sv);
  }
}

// agg_gen = sum over branches (every triplet lands in exactly one branch)
__global__ void agg_gen_kernel(const float* __restrict__ agg,
                               float* __restrict__ agg_gen, int E, int NB) {
  size_t n = (size_t)E * 64;
  size_t i = (size_t)blockIdx.x * blockDim.x + threadIdx.x;
  if (i >= n) return;
  float s = 0.0f;
  for (int b = 0; b < NB; ++b) s += agg[(size_t)b * n + i];
  agg_gen[i] = s;
}

// Main block: 64 rows per block (four M-tiles per wave sharing every B
// fragment -> 4x weight reuse), loop 7 jobs (6 bond branches + generic),
// full 7-GEMM silu/residual chain per job, accumulate coeff*h in registers;
// single store, no output atomics.
__global__ void __launch_bounds__(256)
interaction_kernel(const bf16_t* __restrict__ xbf, const float* __restrict__ agg,
                   const float* __restrict__ agg_gen,
                   const bf16_t* __restrict__ pk_ji, const bf16_t* __restrict__ pk_up,
                   const bf16_t* __restrict__ pk_bs1, const bf16_t* __restrict__ pk_bs2,
                   const bf16_t* __restrict__ pk_lin, const bf16_t* __restrict__ pk_as1,
                   const bf16_t* __restrict__ pk_as2,
                   const float* __restrict__ b_ji, const float* __restrict__ b_bs1,
                   const float* __restrict__ b_bs2, const float* __restrict__ b_lin,
                   const float* __restrict__ b_as1, const float* __restrict__ b_as2,
                   const float* __restrict__ alpha_p, float* __restrict__ out,
                   int E, int NB) {
  constexpr int MT = 4;          // M-tiles per block
  constexpr int MR = MT * 16;    // 64 rows per block
  __shared__ bf16_t xs[MR][136];
  __shared__ bf16_t abuf[MR][136];
  __shared__ bf16_t aggs[MR][72];

  const int tid = threadIdx.x, wv = tid >> 5, lane = tid & 31;
  const int hh = lane >> 4, col = wv * 16 + (lane & 15);
  const int row0 = blockIdx.x * MR;
  const bool full = (row0 + MR <= E);

  if (full) {  // vectorized: 16B per thread per iter
    for (int i = tid; i < MR * 16; i += 256) {
      int r = i >> 4, c8 = (i & 15) * 8;
      *(uint4*)&xs[r][c8] = *(const uint4*)&xbf[(size_t)(row0 + r) * 128 + c8];
    }
  } else {
    for (int i = tid; i < MR * 128; i += 256) {
      int r = i >> 7, c = i & 127;
      int gr = row0 + r;
      xs[r][c] = (gr < E) ? xbf[(size_t)gr * 128 + c] : (bf16_t)0.0f;
    }
  }

  const float alpha = *alpha_p;
  float oacc[MT][8];
#pragma unroll
  for (int m = 0; m < MT; ++m)
#pragma unroll
    for (int r = 0; r < 8; ++r) oacc[m][r] = 0.0f;

  // stage a MRx128 f32-register tile into abuf as bf16 (A operand for next GEMM)
  auto stage = [&](float vals[MT][8]) {
    __syncthreads();
#pragma unroll
    for (int m = 0; m < MT; ++m)
#pragma unroll
      for (int r = 0; r < 8; ++r) abuf[m * 16 + r + 8 * hh][col] = (bf16_t)vals[m][r];
    __syncthreads();
  };

  for (int job = 0; job <= NB; ++job) {
    const int jw = (job < NB) ? job : (NB - 1);
    const float coeff = (job < NB) ? (1.0f - alpha) : alpha;
    const float* asrc = (job < NB) ? (agg + (size_t)job * E * 64) : agg_gen;

    __syncthreads();
    if (full) {  // float4 load -> packed bf16 x4 store
      for (int i = tid; i < MR * 16; i += 256) {
        int r = i >> 4, c4 = (i & 15) * 4;
        float4 v = *(const float4*)&asrc[(size_t)(row0 + r) * 64 + c4];
        bf16_t tmp[4] = {(bf16_t)v.x, (bf16_t)v.y, (bf16_t)v.z, (bf16_t)v.w};
        *(uint2*)&aggs[r][c4] = *(uint2*)tmp;
      }
    } else {
      for (int i = tid; i < MR * 64; i += 256) {
        int r = i >> 6, c = i & 63;
        int gr = row0 + r;
        aggs[r][c] = (gr < E) ? (bf16_t)asrc[(size_t)gr * 64 + c] : (bf16_t)0.0f;
      }
    }
    __syncthreads();

    const size_t w128 = ((size_t)jw * 8 + wv) * 2048;  // KS=4 per-tile block
    const size_t w64  = ((size_t)jw * 8 + wv) * 1024;  // KS=2 per-tile block

    float hv[MT][8], tv[MT][8];
    v8f cc[MT];

    // h = silu(x @ W_ji + b_ji)
#pragma unroll
    for (int m = 0; m < MT; ++m) cc[m] = vzero8();
    wmma_multi<4, MT>(&xs[0][0], 136, pk_ji + w128, lane, cc);
    {
      const float bj = b_ji[jw * 128 + col];
#pragma unroll
      for (int m = 0; m < MT; ++m)
#pragma unroll
        for (int r = 0; r < 8; ++r) hv[m][r] = silu_f(cc[m][r] + bj);
    }
    // h += silu(agg @ W_up)
#pragma unroll
    for (int m = 0; m < MT; ++m) cc[m] = vzero8();
    wmma_multi<2, MT>(&aggs[0][0], 72, pk_up + w64, lane, cc);
#pragma unroll
    for (int m = 0; m < MT; ++m)
#pragma unroll
      for (int r = 0; r < 8; ++r) hv[m][r] += silu_f(cc[m][r]);

    // h += silu(silu(h@W_bs1+b1)@W_bs2+b2)
    stage(hv);
#pragma unroll
    for (int m = 0; m < MT; ++m) cc[m] = vzero8();
    wmma_multi<4, MT>(&abuf[0][0], 136, pk_bs1 + w128, lane, cc);
    {
      const float b1 = b_bs1[jw * 128 + col];
#pragma unroll
      for (int m = 0; m < MT; ++m)
#pragma unroll
        for (int r = 0; r < 8; ++r) tv[m][r] = silu_f(cc[m][r] + b1);
    }
    stage(tv);
#pragma unroll
    for (int m = 0; m < MT; ++m) cc[m] = vzero8();
    wmma_multi<4, MT>(&abuf[0][0], 136, pk_bs2 + w128, lane, cc);
    {
      const float b2 = b_bs2[jw * 128 + col];
#pragma unroll
      for (int m = 0; m < MT; ++m)
#pragma unroll
        for (int r = 0; r < 8; ++r) hv[m][r] += silu_f(cc[m][r] + b2);
    }

    // h = silu(h@W_lin+b) + x
    stage(hv);
#pragma unroll
    for (int m = 0; m < MT; ++m) cc[m] = vzero8();
    wmma_multi<4, MT>(&abuf[0][0], 136, pk_lin + w128, lane, cc);
    {
      const float bl = b_lin[jw * 128 + col];
#pragma unroll
      for (int m = 0; m < MT; ++m)
#pragma unroll
        for (int r = 0; r < 8; ++r)
          hv[m][r] = silu_f(cc[m][r] + bl) + (float)xs[m * 16 + r + 8 * hh][col];
    }

    // h += silu(silu(h@W_as1+b1)@W_as2+b2)
    stage(hv);
#pragma unroll
    for (int m = 0; m < MT; ++m) cc[m] = vzero8();
    wmma_multi<4, MT>(&abuf[0][0], 136, pk_as1 + w128, lane, cc);
    {
      const float ba1 = b_as1[jw * 128 + col];
#pragma unroll
      for (int m = 0; m < MT; ++m)
#pragma unroll
        for (int r = 0; r < 8; ++r) tv[m][r] = silu_f(cc[m][r] + ba1);
    }
    stage(tv);
#pragma unroll
    for (int m = 0; m < MT; ++m) cc[m] = vzero8();
    wmma_multi<4, MT>(&abuf[0][0], 136, pk_as2 + w128, lane, cc);
    {
      const float ba2 = b_as2[jw * 128 + col];
#pragma unroll
      for (int m = 0; m < MT; ++m)
#pragma unroll
        for (int r = 0; r < 8; ++r) hv[m][r] += silu_f(cc[m][r] + ba2);
    }

#pragma unroll
    for (int m = 0; m < MT; ++m)
#pragma unroll
      for (int r = 0; r < 8; ++r) oacc[m][r] += coeff * hv[m][r];
  }

  if (full) {
#pragma unroll
    for (int m = 0; m < MT; ++m)
#pragma unroll
      for (int r = 0; r < 8; ++r)
        out[(size_t)(row0 + m * 16 + r + 8 * hh) * 128 + col] = oacc[m][r];
  } else {
#pragma unroll
    for (int m = 0; m < MT; ++m)
#pragma unroll
      for (int r = 0; r < 8; ++r) {
        int gr = row0 + m * 16 + r + 8 * hh;
        if (gr < E) out[(size_t)gr * 128 + col] = oacc[m][r];
      }
  }
}

static inline unsigned cdiv_u(size_t n, int t) { return (unsigned)((n + t - 1) / t); }

extern "C" void kernel_launch(void* const* d_in, const int* in_sizes, int n_in,
                              void* d_out, int out_size, void* d_ws, size_t ws_size,
                              hipStream_t stream) {
  (void)out_size; (void)ws_size;
  if (n_in < 28) return;
  const float* x      = (const float*)d_in[0];
  const float* rbf    = (const float*)d_in[1];
  const float* sbf    = (const float*)d_in[2];
  const int*   idx_kj = (const int*)d_in[3];
  const int*   idx_ji = (const int*)d_in[4];
  const int*   bt     = (const int*)d_in[5];
  const float* alpha  = (const float*)d_in[6];
  const float* W_rbf1 = (const float*)d_in[8];
  const float* W_rbf2 = (const float*)d_in[9];
  const float* W_sbf1 = (const float*)d_in[10];
  const float* W_sbf2 = (const float*)d_in[11];
  const float* W_kj   = (const float*)d_in[12];
  const float* b_kj   = (const float*)d_in[13];
  const float* W_ji   = (const float*)d_in[14];
  const float* b_ji   = (const float*)d_in[15];
  const float* W_down = (const float*)d_in[16];
  const float* W_up   = (const float*)d_in[17];
  const float* W_bs1  = (const float*)d_in[18];
  const float* b_bs1  = (const float*)d_in[19];
  const float* W_bs2  = (const float*)d_in[20];
  const float* b_bs2  = (const float*)d_in[21];
  const float* W_lin  = (const float*)d_in[22];
  const float* b_lin  = (const float*)d_in[23];
  const float* W_as1  = (const float*)d_in[24];
  const float* b_as1  = (const float*)d_in[25];
  const float* W_as2  = (const float*)d_in[26];
  const float* b_as2  = (const float*)d_in[27];

  const int H = 128, I = 64;
  const int E  = in_sizes[0] / H;
  const int T  = in_sizes[3];
  const int NB = in_sizes[14] / (H * H);
  const int R  = in_sizes[1] / E;
  const int SR = in_sizes[2] / T;
  const int B  = in_sizes[8] / R;
  const int B2 = in_sizes[10] / SR;

  char* ws = (char*)d_ws;
  size_t off = 0;
  auto alloc = [&](size_t bytes) -> void* {
    off = (off + 255) & ~(size_t)255;
    void* p = ws + off;
    off += bytes;
    return p;
  };
  bf16_t* xbf    = (bf16_t*)alloc((size_t)E * H * 2);
  float*  rproj  = (float*) alloc((size_t)E * H * 4);
  float*  xkj    = (float*) alloc((size_t)E * I * 4);
  float*  agg    = (float*) alloc((size_t)NB * E * I * 4);
  float*  agggen = (float*) alloc((size_t)E * I * 4);
  bf16_t* pk_kj  = (bf16_t*)alloc((size_t)H * H * 2);
  bf16_t* pk_dn  = (bf16_t*)alloc((size_t)H * I * 2);
  bf16_t* pk_ji  = (bf16_t*)alloc((size_t)NB * H * H * 2);
  bf16_t* pk_up  = (bf16_t*)alloc((size_t)NB * I * H * 2);
  bf16_t* pk_b1  = (bf16_t*)alloc((size_t)NB * H * H * 2);
  bf16_t* pk_b2  = (bf16_t*)alloc((size_t)NB * H * H * 2);
  bf16_t* pk_li  = (bf16_t*)alloc((size_t)NB * H * H * 2);
  bf16_t* pk_a1  = (bf16_t*)alloc((size_t)NB * H * H * 2);
  bf16_t* pk_a2  = (bf16_t*)alloc((size_t)NB * H * H * 2);

  hipMemsetAsync(agg, 0, (size_t)NB * E * I * 4, stream);

  const int thr = 256;
  pack_w<<<cdiv_u((size_t)H * H, thr), thr, 0, stream>>>(W_kj, pk_kj, H, H, 1);
  pack_w<<<cdiv_u((size_t)H * I, thr), thr, 0, stream>>>(W_down, pk_dn, H, I, 1);
  pack_w<<<cdiv_u((size_t)NB * H * H, thr), thr, 0, stream>>>(W_ji, pk_ji, H, H, NB);
  pack_w<<<cdiv_u((size_t)NB * I * H, thr), thr, 0, stream>>>(W_up, pk_up, I, H, NB);
  pack_w<<<cdiv_u((size_t)NB * H * H, thr), thr, 0, stream>>>(W_bs1, pk_b1, H, H, NB);
  pack_w<<<cdiv_u((size_t)NB * H * H, thr), thr, 0, stream>>>(W_bs2, pk_b2, H, H, NB);
  pack_w<<<cdiv_u((size_t)NB * H * H, thr), thr, 0, stream>>>(W_lin, pk_li, H, H, NB);
  pack_w<<<cdiv_u((size_t)NB * H * H, thr), thr, 0, stream>>>(W_as1, pk_a1, H, H, NB);
  pack_w<<<cdiv_u((size_t)NB * H * H, thr), thr, 0, stream>>>(W_as2, pk_a2, H, H, NB);

  cvt_bf16<<<cdiv_u((size_t)E * H, thr), thr, 0, stream>>>(x, xbf, (size_t)E * H);
  rbfproj_kernel<<<cdiv_u((size_t)E, thr), thr, 0, stream>>>(rbf, W_rbf1, W_rbf2,
                                                             rproj, E, R, B);

  const unsigned rowBlocks16 = (unsigned)((E + 15) / 16);
  xkj_kernel<<<rowBlocks16, 256, 0, stream>>>(xbf, rproj, pk_kj, pk_dn, b_kj, xkj, E);

  triplet_kernel<<<cdiv_u((size_t)T, thr), thr, 0, stream>>>(
      sbf, W_sbf1, W_sbf2, xkj, idx_kj, idx_ji, bt, agg, T, E, SR, B2);

  agg_gen_kernel<<<cdiv_u((size_t)E * I, thr), thr, 0, stream>>>(agg, agggen, E, NB);

  const unsigned rowBlocks64 = (unsigned)((E + 63) / 64);
  interaction_kernel<<<rowBlocks64, 256, 0, stream>>>(
      xbf, agg, agggen, pk_ji, pk_up, pk_b1, pk_b2, pk_li, pk_a1, pk_a2,
      b_ji, b_bs1, b_bs2, b_lin, b_as1, b_as2, alpha, (float*)d_out, E, NB);
}
